// Decoder_45363444580423
// MI455X (gfx1250) — matrix operand order repaired and checked by reading, not compiled, not run on hardware
//
#include <hip/hip_runtime.h>
#include <hip/hip_bf16.h>
#include <stdint.h>

// B=256, N=512, H=256
#define BB 256
#define NN 512
#define HH 256

typedef __bf16  bf16x16 __attribute__((ext_vector_type(16)));
typedef __bf16  bf16x8  __attribute__((ext_vector_type(8)));
typedef float   f32x8   __attribute__((ext_vector_type(8)));

// Native bf16 converts -> let the compiler emit v_cvt_pk_bf16_f32 etc.
__device__ __forceinline__ __bf16 f2bf(float f) { return (__bf16)f; }
__device__ __forceinline__ float  bf2f(__bf16 b) { return (float)b; }
__device__ __forceinline__ float  sigm(float x) { return 1.0f / (1.0f + __expf(-x)); }

__device__ __forceinline__ f32x8 wmma_bf16(bf16x16 a, bf16x16 b, f32x8 c) {
    return __builtin_amdgcn_wmma_f32_16x16x32_bf16(false, a, false, b, (short)0, c, false, false);
}

// A-fragment (16x32, 16-bit, row-major source): per ISA layout,
// lanes 0-15 hold K={0..7,16..23}, lanes 16-31 hold K={8..15,24..31}.
__device__ __forceinline__ bf16x16 load_a_bf(const __bf16* rowbase, int k0, int hi) {
    bf16x8 lo = *(const bf16x8*)(rowbase + k0 + hi * 8);
    bf16x8 hs = *(const bf16x8*)(rowbase + k0 + hi * 8 + 16);
    bf16x16 r;
#pragma unroll
    for (int i = 0; i < 8; ++i) { r[i] = lo[i]; r[i + 8] = hs[i]; }
    return r;
}
// Same fragment converting from an f32 row-major source on the fly.
__device__ __forceinline__ bf16x16 load_a_f32(const float* rowbase, int k0, int hi) {
    const float* p = rowbase + k0 + hi * 8;
    bf16x16 r;
#pragma unroll
    for (int i = 0; i < 8; ++i) { r[i] = f2bf(p[i]); r[i + 8] = f2bf(p[i + 16]); }
    return r;
}
// B-fragment (32x16, 16-bit): lane = N column; lanes 0-15 hold K={0..7,16..23},
// lanes 16-31 hold K={8..15,24..31} (same K order as A) along K of the row-major
// [N,K] weight matrix (computing out = A * W^T).
__device__ __forceinline__ bf16x16 load_b_bf(const __bf16* colrow, int k0, int hi) {
    bf16x8 lo = *(const bf16x8*)(colrow + k0 + hi * 8);
    bf16x8 hs = *(const bf16x8*)(colrow + k0 + hi * 8 + 16);
    bf16x16 r;
#pragma unroll
    for (int i = 0; i < 8; ++i) { r[i] = lo[i]; r[i + 8] = hs[i]; }
    return r;
}

// ---------------- prep: fold weights/biases, convert to bf16 ----------------
__global__ void prep_kernel(const float* __restrict__ Wih, const float* __restrict__ Whh,
                            const float* __restrict__ bih, const float* __restrict__ bhh,
                            const float* __restrict__ Wq,  const float* __restrict__ Wref,
                            __bf16* __restrict__ WcB, __bf16* __restrict__ WqB,
                            __bf16* __restrict__ WrefB, float* __restrict__ bc) {
    int i = blockIdx.x * blockDim.x + threadIdx.x;
    int stride = gridDim.x * blockDim.x;
    for (int k = i; k < 4 * HH * HH; k += stride) WcB[k] = f2bf(Wih[k] + Whh[k]);
    for (int k = i; k < HH * HH; k += stride)     WqB[k] = f2bf(Wq[k]);
    for (int k = i; k < HH * HH; k += stride)     WrefB[k] = f2bf(Wref[k]);
    for (int k = i; k < 4 * HH; k += stride)      bc[k] = bih[k] + bhh[k];
}

__global__ void init_state_kernel(__bf16* __restrict__ hid0, float* __restrict__ cell) {
    int i = blockIdx.x * blockDim.x + threadIdx.x;
    int stride = gridDim.x * blockDim.x;
    for (int k = i; k < BB * HH; k += stride) { hid0[k] = f2bf(0.0f); cell[k] = 0.0f; }
}

// ---------------- enc_proj = enc @ W_ref^T, stored bf16 (L2-resident) -------
// 4 waves / block, each wave owns one 16-row M tile.
__global__ void encproj_kernel(const float* __restrict__ enc, const __bf16* __restrict__ Wref,
                               __bf16* __restrict__ ep) {
    int wave = threadIdx.x >> 5;
    int lane = threadIdx.x & 31;
    int mt = (blockIdx.x * 4 + wave) * 16;   // row tile over B*N = 131072
    int nt = blockIdx.y * 16;                // col tile over H
    int l16 = lane & 15, hi = lane >> 4;
    const float*  arow = enc  + (size_t)(mt + l16) * HH;
    const __bf16* brow = Wref + (size_t)(nt + l16) * HH;
    f32x8 acc = {};
#pragma unroll
    for (int k0 = 0; k0 < HH; k0 += 32) {
        bf16x16 a = load_a_f32(arow, k0, hi);
        bf16x16 b = load_b_bf(brow, k0, hi);
        acc = wmma_bf16(a, b, acc);
    }
#pragma unroll
    for (int r = 0; r < 8; ++r) {
        size_t m = (size_t)(mt + hi * 8 + r);
        ep[m * HH + nt + l16] = f2bf(acc[r]);
    }
}

// ---------------- fused gates GEMM + LSTM pointwise -------------------------
// 4 waves / block, each wave owns one 16-row batch tile.
__global__ void lstm_step_kernel(const __bf16* __restrict__ hid_in,
                                 const __bf16* __restrict__ Wc,   // [4H,H] bf16
                                 const float*  __restrict__ bc,   // [4H]
                                 float* __restrict__ cell,        // [B,H]
                                 __bf16* __restrict__ hid_out) {  // [B,H]
    int wave = threadIdx.x >> 5;
    int lane = threadIdx.x & 31;
    int bt = (blockIdx.x * 4 + wave) * 16;   // batch tile
    int ht = blockIdx.y * 16;                // hidden tile
    int l16 = lane & 15, hi = lane >> 4;
    const __bf16* arow = hid_in + (size_t)(bt + l16) * HH;
    const __bf16* w0 = Wc + (size_t)(0 * HH + ht + l16) * HH;
    const __bf16* w1 = Wc + (size_t)(1 * HH + ht + l16) * HH;
    const __bf16* w2 = Wc + (size_t)(2 * HH + ht + l16) * HH;
    const __bf16* w3 = Wc + (size_t)(3 * HH + ht + l16) * HH;
    f32x8 acc0 = {}, acc1 = {}, acc2 = {}, acc3 = {};
#pragma unroll
    for (int k0 = 0; k0 < HH; k0 += 32) {
        bf16x16 a = load_a_bf(arow, k0, hi);
        acc0 = wmma_bf16(a, load_b_bf(w0, k0, hi), acc0);
        acc1 = wmma_bf16(a, load_b_bf(w1, k0, hi), acc1);
        acc2 = wmma_bf16(a, load_b_bf(w2, k0, hi), acc2);
        acc3 = wmma_bf16(a, load_b_bf(w3, k0, hi), acc3);
    }
    int n = ht + l16;
    float bi = bc[0 * HH + n], bf = bc[1 * HH + n], bg = bc[2 * HH + n], bo = bc[3 * HH + n];
#pragma unroll
    for (int r = 0; r < 8; ++r) {
        int m = bt + hi * 8 + r;
        size_t idx = (size_t)m * HH + n;
        float ig = sigm(acc0[r] + bi);
        float fg = sigm(acc1[r] + bf);
        float gg = tanhf(acc2[r] + bg);
        float og = sigm(acc3[r] + bo);
        float cn = fg * cell[idx] + ig * gg;
        cell[idx] = cn;
        hid_out[idx] = f2bf(og * tanhf(cn));
    }
}

// ---------------- q = hid @ W_q^T -------------------------------------------
__global__ void q_kernel(const __bf16* __restrict__ hid, const __bf16* __restrict__ Wq,
                         float* __restrict__ q) {
    int wave = threadIdx.x >> 5;
    int lane = threadIdx.x & 31;
    int bt = (blockIdx.x * 4 + wave) * 16;
    int nt = blockIdx.y * 16;
    int l16 = lane & 15, hi = lane >> 4;
    const __bf16* arow = hid + (size_t)(bt + l16) * HH;
    const __bf16* brow = Wq  + (size_t)(nt + l16) * HH;
    f32x8 acc = {};
#pragma unroll
    for (int k0 = 0; k0 < HH; k0 += 32) {
        bf16x16 a = load_a_bf(arow, k0, hi);
        bf16x16 b = load_b_bf(brow, k0, hi);
        acc = wmma_bf16(a, b, acc);
    }
#pragma unroll
    for (int r = 0; r < 8; ++r)
        q[(size_t)(bt + hi * 8 + r) * HH + nt + l16] = acc[r];
}

// ---------------- logits[b,n] = sum_h v[h]*tanh(ep[b,n,h] + q[b,h]) ---------
__global__ void attn_kernel(const __bf16* __restrict__ ep, const float* __restrict__ q,
                            const float* __restrict__ v, float* __restrict__ out, int t) {
    int w = threadIdx.x >> 5;
    int lane = threadIdx.x & 31;
    int p = blockIdx.x * 8 + w;                 // (b,n) pair index in [0, B*N)
    int b = p >> 9, n = p & (NN - 1);
    const __bf16* erow = ep + (size_t)p * HH;
    const float*  qrow = q + (size_t)b * HH;
    int h0 = lane * 8;
    bf16x8 e = *(const bf16x8*)(erow + h0);
    float s = 0.0f;
#pragma unroll
    for (int i = 0; i < 8; ++i)
        s += v[h0 + i] * tanhf(bf2f(e[i]) + qrow[h0 + i]);
#pragma unroll
    for (int off = 16; off > 0; off >>= 1) s += __shfl_xor(s, off, 32);
    if (lane == 0) out[(size_t)b * (NN * NN) + (size_t)t * NN + n] = s;
}

extern "C" void kernel_launch(void* const* d_in, const int* in_sizes, int n_in,
                              void* d_out, int out_size, void* d_ws, size_t ws_size,
                              hipStream_t stream) {
    (void)in_sizes; (void)n_in; (void)out_size; (void)ws_size;
    const float* enc  = (const float*)d_in[0];
    const float* Wih  = (const float*)d_in[1];
    const float* Whh  = (const float*)d_in[2];
    const float* bih  = (const float*)d_in[3];
    const float* bhh  = (const float*)d_in[4];
    const float* Wref = (const float*)d_in[5];
    const float* Wq   = (const float*)d_in[6];
    const float* v    = (const float*)d_in[7];
    float* out = (float*)d_out;

    char* ws = (char*)d_ws;
    size_t off = 0;
    auto carve = [&](size_t bytes) { void* p = ws + off; off = (off + bytes + 255) & ~(size_t)255; return p; };
    __bf16* ep    = (__bf16*)carve((size_t)BB * NN * HH * 2);   // 64 MiB, L2-resident
    __bf16* WcB   = (__bf16*)carve((size_t)4 * HH * HH * 2);
    __bf16* WqB   = (__bf16*)carve((size_t)HH * HH * 2);
    __bf16* WrefB = (__bf16*)carve((size_t)HH * HH * 2);
    float*  bc    = (float*) carve((size_t)4 * HH * 4);
    __bf16* hid0  = (__bf16*)carve((size_t)BB * HH * 2);
    __bf16* hid1  = (__bf16*)carve((size_t)BB * HH * 2);
    float*  cell  = (float*) carve((size_t)BB * HH * 4);
    float*  q     = (float*) carve((size_t)BB * HH * 4);
    __bf16* hidb[2] = { hid0, hid1 };

    prep_kernel<<<512, 256, 0, stream>>>(Wih, Whh, bih, bhh, Wq, Wref, WcB, WqB, WrefB, bc);
    init_state_kernel<<<256, 256, 0, stream>>>(hid0, cell);

    // enc_proj: M = B*N = 131072 rows; 4 waves/block, 16x16 tile per wave
    encproj_kernel<<<dim3((BB * NN) / 64, HH / 16), 128, 0, stream>>>(enc, WrefB, ep);

    for (int t = 0; t < NN; ++t) {
        const __bf16* hin = hidb[t & 1];
        __bf16* hout = hidb[(t + 1) & 1];
        lstm_step_kernel<<<dim3(BB / 64, HH / 16), 128, 0, stream>>>(hin, WcB, bc, cell, hout);
        q_kernel<<<dim3(BB / 64, HH / 16), 128, 0, stream>>>(hout, WqB, q);
        attn_kernel<<<(BB * NN) / 8, 256, 0, stream>>>(ep, q, v, out, t);
    }
}
